// SingleRenderer_88115549045093
// MI455X (gfx1250) — matrix operand order repaired and checked by reading, not compiled, not run on hardware
//
#include <hip/hip_runtime.h>
#include <hip/hip_bf16.h>

typedef __attribute__((ext_vector_type(16))) _Float16 v16h;
typedef __attribute__((ext_vector_type(8)))  float    v8f;

#define NUM_RAYS 16384
#define RPB 16          // rays per block
#define THREADS 128     // 4 wave32 waves
#define NW 4

#define LOG2E_F 1.4426950408889634f
#define LN2_F   0.6931471805599453f

// raw hardware transcendentals: v_exp_f32 / v_log_f32 / v_rcp_f32 (no libm fixup code)
__device__ __forceinline__ float fexp2_(float x){ return __builtin_amdgcn_exp2f(x); }
__device__ __forceinline__ float flog2_(float x){ return __builtin_amdgcn_logf(x); }
__device__ __forceinline__ float frcp_(float x){ return __builtin_amdgcn_rcpf(x); }

__device__ __forceinline__ float sigmoidf_(float x){
  // 1/(1+exp(-x)) ; arg of log2-domain exp; 1+e >= 1 so rcp is safe
  return frcp_(1.0f + fexp2_(-x * LOG2E_F));
}
__device__ __forceinline__ float softplusf_(float x){
  // ln(1+exp(x)) = ln2 * log2(1 + 2^(x*log2e)); 1+e >= 1 so v_log_f32 needs no denorm rescue
  float sp = LN2_F * flog2_(1.0f + fexp2_(x * LOG2E_F));
  return (x > 20.0f) ? x : sp;
}

__device__ __forceinline__ v8f wmma16(v16h a, v16h b, v8f c){
  // D = A(16x32 f16) * B(32x16 f16) + C(16x16 f32)
  return __builtin_amdgcn_wmma_f32_16x16x32_f16(false, a, false, b, (short)0, c, false, false);
}

__global__ __launch_bounds__(THREADS)
void neus_render_kernel(
    const float* __restrict__ rays_o, const float* __restrict__ rays_d,
    const float* __restrict__ nearp,  const float* __restrict__ farp,
    const float* __restrict__ s_scalar,
    const float* __restrict__ w1, const float* __restrict__ b1,
    const float* __restrict__ w2, const float* __restrict__ b2,
    const float* __restrict__ w3, const float* __restrict__ b3,
    const float* __restrict__ rw1, const float* __restrict__ rb1,
    const float* __restrict__ rw2, const float* __restrict__ rb2,
    float* __restrict__ out)
{
  __shared__ float s_d[RPB][128];        // sample depths (grows 64 -> 128)
  __shared__ float s_sdf[RPB][128];      // sdf at samples
  __shared__ float s_scr[RPB][128];      // cdf scratch during upsample; vis_w at the end
  __shared__ float s_dfine[RPB][16];
  __shared__ float s_sdffine[RPB][16];
  __shared__ float s_rgb[RPB][4];
  __shared__ float s_o[RPB][3], s_dir[RPB][3];
  __shared__ _Float16 s_w1[32*64];       // K padded 3 -> 32
  __shared__ _Float16 s_w2[64*64];
  __shared__ _Float16 s_w3[64*16];       // N padded 1 -> 16
  __shared__ _Float16 s_rw1[32*64];      // K padded 6 -> 32
  __shared__ _Float16 s_rw2[64*16];      // N padded 3 -> 16
  __shared__ float s_b1[64], s_b2[64], s_rb1[64], s_rb2v[3];
  __shared__ float s_b3f, s_sval;
  __shared__ _Float16 s_h[NW][16*64];    // per-wave hidden activation tile
  __shared__ float s_norm[NW][16];       // |pt| per tile row

  const int tid  = threadIdx.x;
  const int lane = tid & 31;
  const int wv   = tid >> 5;
  const int half = lane >> 4;
  const int mrow = lane & 15;
  const int ray0 = blockIdx.x * RPB;

  // ---------------- weights -> LDS (f16, padded) ----------------
  for (int i = tid; i < 32*64; i += THREADS){ int k=i>>6, n=i&63; s_w1[i]  = (_Float16)(k<3 ? w1[k*64+n]  : 0.f); }
  for (int i = tid; i < 64*64; i += THREADS){                      s_w2[i]  = (_Float16)w2[i]; }
  for (int i = tid; i < 64*16; i += THREADS){ int k=i>>4, n=i&15; s_w3[i]  = (_Float16)(n==0 ? w3[k]      : 0.f); }
  for (int i = tid; i < 32*64; i += THREADS){ int k=i>>6, n=i&63; s_rw1[i] = (_Float16)(k<6 ? rw1[k*64+n] : 0.f); }
  for (int i = tid; i < 64*16; i += THREADS){ int k=i>>4, n=i&15; s_rw2[i] = (_Float16)(n<3 ? rw2[k*3+n]  : 0.f); }
  if (tid < 64){ s_b1[tid]=b1[tid]; s_b2[tid]=b2[tid]; s_rb1[tid]=rb1[tid]; }
  if (tid < 3)  s_rb2v[tid]=rb2[tid];
  if (tid == 0){ s_b3f = b3[0]; s_sval = s_scalar[0]; }
  if (tid < RPB){
    int r = ray0 + tid;
    float dx=rays_d[r*3], dy=rays_d[r*3+1], dz=rays_d[r*3+2];
    float inv = __builtin_amdgcn_rsqf(dx*dx+dy*dy+dz*dz);
    s_dir[tid][0]=dx*inv; s_dir[tid][1]=dy*inv; s_dir[tid][2]=dz*inv;
    s_o[tid][0]=rays_o[r*3]; s_o[tid][1]=rays_o[r*3+1]; s_o[tid][2]=rays_o[r*3+2];
  }
  if (tid < RPB*4) s_rgb[tid>>2][tid&3] = 0.f;
  // coarse depths: near*(1-t)+far*t, t = j/63
  for (int i = tid; i < RPB*64; i += THREADS){
    int r = i >> 6, j = i & 63;
    float t = (float)j * (1.0f/63.0f);
    s_d[r][j] = nearp[ray0+r]*(1.f-t) + farp[ray0+r]*t;
  }
  __syncthreads();

  // ---------------- loop-invariant B fragments in registers ----------------
  // B 32x16 f16 as loaded here (contiguous k, same order as A): lanes 0-15 = col n, K 0..15 (pairs per VGPR); lanes 16-31 = K 16..31
  auto loadB = [&](const _Float16* w, int stride, int kbase, int nbase) -> v16h {
    v16h bf = {};
    int ncol = nbase + mrow;
    #pragma unroll
    for (int i = 0; i < 8; ++i){
      int k = kbase + i*2 + half*16;
      bf[2*i]   = w[k*stride + ncol];
      bf[2*i+1] = w[(k+1)*stride + ncol];
    }
    return bf;
  };
  v16h w1B[4], w2B[2][4], w3B[2], rw1B[4], rw2B[2];
  #pragma unroll
  for (int nt=0; nt<4; ++nt) w1B[nt] = loadB(s_w1, 64, 0, nt*16);
  #pragma unroll
  for (int kt=0; kt<2; ++kt)
    #pragma unroll
    for (int nt=0; nt<4; ++nt) w2B[kt][nt] = loadB(s_w2, 64, kt*32, nt*16);
  #pragma unroll
  for (int kt=0; kt<2; ++kt) w3B[kt]  = loadB(s_w3, 16, kt*32, 0);
  #pragma unroll
  for (int nt=0; nt<4; ++nt) rw1B[nt] = loadB(s_rw1, 64, 0, nt*16);
  #pragma unroll
  for (int kt=0; kt<2; ++kt) rw2B[kt] = loadB(s_rw2, 16, kt*32, 0);

  // A 16x32 f16 as loaded here (contiguous k, same order as B): lane half=0 holds K 0..15, half=1 holds K 16..31
  auto loadA_h = [&](int kbase) -> v16h {
    v16h af = {};
    const _Float16* hb = s_h[wv];
    #pragma unroll
    for (int i = 0; i < 8; ++i){
      int k = kbase + i*2 + half*16;
      af[2*i]   = hb[mrow*64 + k];
      af[2*i+1] = hb[mrow*64 + k + 1];
    }
    return af;
  };

  // ---------------- sdf MLP on one 16-point tile (wave-scope) ----------------
  auto sdf_tile = [&](int base, const float* dsrc, int dstride,
                      float* outb, int ostride, int div){
    int rowg = base + mrow;
    int ray = rowg / div, j = rowg - ray*div;
    float dv = dsrc[ray*dstride + j];
    float px = s_o[ray][0] + dv*s_dir[ray][0];
    float py = s_o[ray][1] + dv*s_dir[ray][1];
    float pz = s_o[ray][2] + dv*s_dir[ray][2];
    if (half == 0) s_norm[wv][mrow] = __builtin_amdgcn_sqrtf(px*px+py*py+pz*pz);
    v16h a1 = {};
    if (half == 0){ a1[0]=(_Float16)px; a1[1]=(_Float16)py; a1[2]=(_Float16)pz; }
    // layer 1 (K padded to 32)
    #pragma unroll
    for (int nt=0; nt<4; ++nt){
      v8f c = {};
      c = wmma16(a1, w1B[nt], c);
      float bias = s_b1[nt*16 + mrow];
      #pragma unroll
      for (int r=0;r<8;++r){
        int m = r + half*8;
        s_h[wv][m*64 + nt*16 + mrow] = (_Float16)softplusf_(c[r] + bias);
      }
    }
    asm volatile("s_wait_dscnt 0" ::: "memory");
    // layer 2 (64x64)
    v16h a20 = loadA_h(0), a21 = loadA_h(32);
    float h2[4][8];
    #pragma unroll
    for (int nt=0; nt<4; ++nt){
      v8f c = {};
      c = wmma16(a20, w2B[0][nt], c);
      c = wmma16(a21, w2B[1][nt], c);
      float bias = s_b2[nt*16 + mrow];
      #pragma unroll
      for (int r=0;r<8;++r) h2[nt][r] = softplusf_(c[r] + bias);
    }
    asm volatile("s_wait_dscnt 0" ::: "memory");
    #pragma unroll
    for (int nt=0; nt<4; ++nt)
      #pragma unroll
      for (int r=0;r<8;++r){
        int m = r + half*8;
        s_h[wv][m*64 + nt*16 + mrow] = (_Float16)h2[nt][r];
      }
    asm volatile("s_wait_dscnt 0" ::: "memory");
    // layer 3 (64 -> 1, N padded to 16)
    v16h a30 = loadA_h(0), a31 = loadA_h(32);
    v8f c3 = {};
    c3 = wmma16(a30, w3B[0], c3);
    c3 = wmma16(a31, w3B[1], c3);
    if (mrow == 0){
      #pragma unroll
      for (int r=0;r<8;++r){
        int m = r + half*8;
        int rg = base + m;
        int ry = rg / div, jj = rg - ry*div;
        outb[ry*ostride + jj] = s_norm[wv][m] - 0.8f + 0.1f*(c3[r] + s_b3f);
      }
    }
  };

  // ---------------- radiance MLP on one 16-midpoint tile ----------------
  auto rad_tile = [&](int base){
    int rowg = base + mrow;
    int ray = rowg / 127, j = rowg - ray*127;
    float dm = 0.5f*(s_d[ray][j] + s_d[ray][j+1]);
    float ddx = s_dir[ray][0], ddy = s_dir[ray][1], ddz = s_dir[ray][2];
    float px = s_o[ray][0] + dm*ddx;
    float py = s_o[ray][1] + dm*ddy;
    float pz = s_o[ray][2] + dm*ddz;
    v16h a1 = {};
    if (half == 0){
      a1[0]=(_Float16)px;  a1[1]=(_Float16)py;  a1[2]=(_Float16)pz;
      a1[3]=(_Float16)ddx; a1[4]=(_Float16)ddy; a1[5]=(_Float16)ddz;
    }
    #pragma unroll
    for (int nt=0; nt<4; ++nt){
      v8f c = {};
      c = wmma16(a1, rw1B[nt], c);
      float bias = s_rb1[nt*16 + mrow];
      #pragma unroll
      for (int r=0;r<8;++r){
        int m = r + half*8;
        s_h[wv][m*64 + nt*16 + mrow] = (_Float16)softplusf_(c[r] + bias);
      }
    }
    asm volatile("s_wait_dscnt 0" ::: "memory");
    v16h a20 = loadA_h(0), a21 = loadA_h(32);
    v8f c2 = {};
    c2 = wmma16(a20, rw2B[0], c2);
    c2 = wmma16(a21, rw2B[1], c2);
    if (mrow < 3){
      #pragma unroll
      for (int r=0;r<8;++r){
        int m = r + half*8;
        int rg = base + m;
        int ry = rg / 127, jj = rg - ry*127;
        float val = sigmoidf_(c2[r] + s_rb2v[mrow]);
        atomicAdd(&s_rgb[ry][mrow], s_scr[ry][jj] * val);  // ds_add_f32
      }
    }
  };

  // ================= coarse sdf =================
  for (int tb = wv; tb < (RPB*64)/16; tb += NW)
    sdf_tile(tb*16, &s_d[0][0], 128, &s_sdf[0][0], 128, 64);
  __syncthreads();

  // ================= 4 upsampling iterations =================
  int n = 64;
  for (int it = 0; it < 4; ++it){
    if (tid < RPB){
      int ry = tid;
      float si = 64.0f * (float)(1 << it);
      float cum = 1.0f, wsum = 0.0f, prevdot = 0.0f;
      for (int j = 0; j < n-1; ++j){
        float sp = s_sdf[ry][j], sn = s_sdf[ry][j+1];
        float zp = s_d[ry][j],   zn = s_d[ry][j+1];
        float mid = 0.5f*(sp+sn);
        float dt  = (sn-sp)/(zn-zp+1e-5f);
        float dvv = fminf((j==0)?0.0f:prevdot, dt);
        prevdot = dt;
        dvv = fminf(fmaxf(dvv, -10.0f), 0.0f);
        float dist = zn - zp;
        float pc = sigmoidf_((mid - dvv*dist*0.5f)*si);
        float nc = sigmoidf_((mid + dvv*dist*0.5f)*si);
        float alpha = (pc - nc + 1e-5f)/(pc + 1e-5f);
        float w = alpha*cum + 1e-5f;            // _alpha_to_w then +1e-5 in sample_pdf
        cum *= (1.0f - alpha + 1e-10f);
        s_scr[ry][j] = w; wsum += w;
      }
      float inv = frcp_(wsum), acc = 0.0f;
      for (int j = 0; j < n; ++j){              // in-place cdf = [0, cumsum(pdf)]
        float wj = (j < n-1) ? s_scr[ry][j] : 0.0f;
        s_scr[ry][j] = acc;
        acc += wj*inv;
      }
      int ind = 0;
      for (int jj = 0; jj < 16; ++jj){          // inverse-CDF sampling, u monotone
        float u = (float)jj * (1.0f/15.0f);
        while (ind < n && s_scr[ry][ind] <= u) ++ind;
        int below = ind-1; below = below < 0 ? 0 : (below > n-1 ? n-1 : below);
        int above = ind > n-1 ? n-1 : ind;
        float cb = s_scr[ry][below], ca = s_scr[ry][above];
        float bb = s_d[ry][below],   ba = s_d[ry][above];
        float den = ca - cb; if (den < 1e-5f) den = 1.0f;
        s_dfine[ry][jj] = bb + ((u - cb)*frcp_(den))*(ba - bb);
      }
    }
    __syncthreads();
    for (int tb = wv; tb < (RPB*16)/16; tb += NW)
      sdf_tile(tb*16, &s_dfine[0][0], 16, &s_sdffine[0][0], 16, 16);
    __syncthreads();
    if (tid < RPB){                             // in-place backward merge of sorted lists
      int ry = tid;
      int i2 = n-1, j2 = 15, k2 = n+15;
      while (j2 >= 0){
        if (i2 >= 0 && s_d[ry][i2] > s_dfine[ry][j2]){
          s_d[ry][k2] = s_d[ry][i2]; s_sdf[ry][k2] = s_sdf[ry][i2]; --i2;
        } else {
          s_d[ry][k2] = s_dfine[ry][j2]; s_sdf[ry][k2] = s_sdffine[ry][j2]; --j2;
        }
        --k2;
      }
    }
    n += 16;
    __syncthreads();
  }

  // ================= final sdf over all 128 samples =================
  for (int tb = wv; tb < (RPB*128)/16; tb += NW)
    sdf_tile(tb*16, &s_d[0][0], 128, &s_sdf[0][0], 128, 128);
  __syncthreads();

  // ================= visibility weights =================
  if (tid < RPB){
    int ry = tid;
    float sv = s_sval;
    float cum = 1.0f;
    float cprev = sigmoidf_(s_sdf[ry][0]*sv);
    for (int j = 0; j < 127; ++j){
      float cnext = sigmoidf_(s_sdf[ry][j+1]*sv);
      float alpha = fmaxf((cprev - cnext)*frcp_(cprev + 1e-10f), 0.0f);
      s_scr[ry][j] = alpha*cum;
      cum *= (1.0f - alpha + 1e-10f);
      cprev = cnext;
    }
  }
  __syncthreads();

  // ================= radiance + accumulate =================
  for (int tb = wv; tb < (RPB*127)/16; tb += NW)
    rad_tile(tb*16);
  __syncthreads();

  if (tid < RPB*3){
    int ry = tid/3, ch = tid - ry*3;
    out[(ray0+ry)*3 + ch] = s_rgb[ry][ch];
  }
}

extern "C" void kernel_launch(void* const* d_in, const int* in_sizes, int n_in,
                              void* d_out, int out_size, void* d_ws, size_t ws_size,
                              hipStream_t stream) {
  (void)in_sizes; (void)n_in; (void)d_ws; (void)ws_size; (void)out_size;
  const float* rays_o = (const float*)d_in[0];
  const float* rays_d = (const float*)d_in[1];
  const float* nearp  = (const float*)d_in[2];
  const float* farp   = (const float*)d_in[3];
  const float* s_sc   = (const float*)d_in[4];
  const float* w1     = (const float*)d_in[5];
  const float* b1     = (const float*)d_in[6];
  const float* w2     = (const float*)d_in[7];
  const float* b2     = (const float*)d_in[8];
  const float* w3     = (const float*)d_in[9];
  const float* b3     = (const float*)d_in[10];
  const float* rw1    = (const float*)d_in[11];
  const float* rb1    = (const float*)d_in[12];
  const float* rw2    = (const float*)d_in[13];
  const float* rb2    = (const float*)d_in[14];
  float* out = (float*)d_out;

  neus_render_kernel<<<NUM_RAYS/RPB, THREADS, 0, stream>>>(
      rays_o, rays_d, nearp, farp, s_sc,
      w1, b1, w2, b2, w3, b3, rw1, rb1, rw2, rb2, out);
}
